// UnifiedModalEncoder_37623913513505
// MI455X (gfx1250) — matrix operand order repaired and checked by reading, not compiled, not run on hardware
//
#include <hip/hip_runtime.h>

typedef __attribute__((ext_vector_type(16))) __bf16 v16bf;
typedef __attribute__((ext_vector_type(8)))  float  v8f;

#define DIM_D 1024
#define DIM_F 4096
#define NE    8
#define NL    6
#define NTOK  2048   /* B*S = 8*256 */

// gfx1250 async global->LDS copy path (ASYNCcnt-tracked, ISA 15.18.3)
#define USE_ASYNC_LDS 1

// ---------------------------------------------------------------- utilities
__global__ __launch_bounds__(256) void zero_kernel(float* p, int n) {
  int i = blockIdx.x * blockDim.x + threadIdx.x;
  if (i < n) p[i] = 0.0f;
}

__global__ __launch_bounds__(256) void copy4_kernel(const float4* __restrict__ a,
                                                    float4* __restrict__ b, int n4) {
  int i = blockIdx.x * blockDim.x + threadIdx.x;
  if (i < n4) b[i] = a[i];
}

__global__ __launch_bounds__(256) void add4_kernel(float4* __restrict__ x,
                                                   const float4* __restrict__ a, int n4) {
  int i = blockIdx.x * blockDim.x + threadIdx.x;
  if (i < n4) {
    float4 v = x[i], w = a[i];
    v.x += w.x; v.y += w.y; v.z += w.z; v.w += w.w;
    x[i] = v;
  }
}

// ---------------------------------------------------------------- layernorm (fp32 -> bf16)
__global__ __launch_bounds__(256) void ln_kernel(const float* __restrict__ x,
                                                 __bf16* __restrict__ xn) {
  __shared__ float red[256];
  const int t = blockIdx.x;                       // token
  const float* row = x + (size_t)t * DIM_D;
  float s = 0.f, s2 = 0.f;
  for (int i = threadIdx.x; i < DIM_D; i += 256) { float v = row[i]; s += v; s2 += v * v; }
  red[threadIdx.x] = s; __syncthreads();
  for (int o = 128; o > 0; o >>= 1) { if (threadIdx.x < o) red[threadIdx.x] += red[threadIdx.x + o]; __syncthreads(); }
  const float mean = red[0] * (1.0f / DIM_D);
  __syncthreads();
  red[threadIdx.x] = s2; __syncthreads();
  for (int o = 128; o > 0; o >>= 1) { if (threadIdx.x < o) red[threadIdx.x] += red[threadIdx.x + o]; __syncthreads(); }
  const float var = red[0] * (1.0f / DIM_D) - mean * mean;
  const float inv = rsqrtf(var + 1e-5f);
  __bf16* orow = xn + (size_t)t * DIM_D;
  for (int i = threadIdx.x; i < DIM_D; i += 256)
    orow[i] = (__bf16)((row[i] - mean) * inv);
}

// ---------------------------------------------------------------- router: wave32 per token
__global__ __launch_bounds__(256) void router_kernel(const __bf16* __restrict__ xn,
                                                     const float* __restrict__ wr,   // [D, E]
                                                     float* __restrict__ gates,      // [NTOK, NE]
                                                     float* __restrict__ stats) {    // [16]: cnt, sump
  const int gtid = blockIdx.x * blockDim.x + threadIdx.x;
  const int token = gtid >> 5;
  const int lane  = gtid & 31;
  if (token >= NTOK) return;
  float acc[NE];
#pragma unroll
  for (int e = 0; e < NE; ++e) acc[e] = 0.f;
  const __bf16* xr = xn + (size_t)token * DIM_D;
  for (int d = lane; d < DIM_D; d += 32) {
    const float xv = (float)xr[d];
    const float* w = wr + (size_t)d * NE;
#pragma unroll
    for (int e = 0; e < NE; ++e) acc[e] += xv * w[e];
  }
#pragma unroll
  for (int e = 0; e < NE; ++e)
    for (int off = 16; off > 0; off >>= 1)
      acc[e] += __shfl_xor(acc[e], off, 32);
  if (lane == 0) {
    float mx = acc[0];
    for (int e = 1; e < NE; ++e) mx = fmaxf(mx, acc[e]);
    float p[NE]; float s = 0.f;
    for (int e = 0; e < NE; ++e) {
      p[e] = __builtin_amdgcn_exp2f((acc[e] - mx) * 1.4426950408889634f);
      s += p[e];
    }
    const float invs = __builtin_amdgcn_rcpf(s);
    for (int e = 0; e < NE; ++e) p[e] *= invs;
    int i1 = 0;
    for (int e = 1; e < NE; ++e) if (p[e] > p[i1]) i1 = e;
    int i2 = -1;
    for (int e = 0; e < NE; ++e) { if (e == i1) continue; if (i2 < 0 || p[e] > p[i2]) i2 = e; }
    float* g = gates + (size_t)token * NE;
    for (int e = 0; e < NE; ++e) g[e] = (e == i1 || e == i2) ? p[e] : 0.f;
    atomicAdd(&stats[i1], 1.0f);
    atomicAdd(&stats[i2], 1.0f);
    for (int e = 0; e < NE; ++e) atomicAdd(&stats[8 + e], p[e]);
  }
}

__global__ void loss_kernel(const float* __restrict__ stats, float* __restrict__ loss) {
  if (threadIdx.x == 0 && blockIdx.x == 0) {
    float s = 0.f;
    for (int e = 0; e < NE; ++e)
      s += (stats[e] * (1.0f / NTOK)) * (stats[8 + e] * (1.0f / NTOK));
    *loss += (float)NE * s;
  }
}

// branchless tanh-approx GELU via v_exp_f32
__device__ __forceinline__ float gelu_f(float v) {
  const float t = 0.7978845608028654f * (v + 0.044715f * v * v * v);
  const float a = fabsf(t);
  const float e = __builtin_amdgcn_exp2f(-2.8853900817779268f * a);
  float th = (1.f - e) * __builtin_amdgcn_rcpf(1.f + e);
  th = copysignf(th, t);
  return 0.5f * v * (1.f + th);
}

// ---------------------------------------------------------------- WMMA GEMM
// C[M,N] = A[M,K](bf16) * Bw[K,N](fp32, converted to bf16 in the tile loader)
// MODE 1: H = gelu(C + bias)  (bf16 out)
// MODE 2: Acc (=/+=) gate[m,expert] * (C + bias)   (fp32 out)
template <int MODE>
__global__ __launch_bounds__(256) void gemm_kernel(
    const __bf16* __restrict__ A, const float* __restrict__ Bw,
    const float* __restrict__ bias, const float* __restrict__ gate, int expert,
    __bf16* __restrict__ Hout, float* __restrict__ Acc, int M, int N, int K) {
  constexpr int BM = 128, BN = 128, BK = 64;
  __shared__ __attribute__((aligned(16))) __bf16 As[2][BM][BK];   // row-major (M,K)
  __shared__ __attribute__((aligned(16))) __bf16 Bs[2][BN][BK];   // transposed (N,K)

  const int tid  = threadIdx.x;
  const int wave = tid >> 5;
  const int lane = tid & 31;
  const int m0 = blockIdx.y * BM;
  const int n0 = blockIdx.x * BN;
  const int r = lane & 15;        // A row / B col within 16
  const int h = lane >> 4;        // lane-half selects K sub-range

  v8f c[8];
#pragma unroll
  for (int i = 0; i < 8; ++i)
#pragma unroll
    for (int j = 0; j < 8; ++j) c[i][j] = 0.f;

  // --- staging helpers -------------------------------------------------
  // A tile: 256 threads x 64B contiguous bf16 copies (async global->LDS)
  auto stageA = [&](int buf, int k0) {
    const int row = tid >> 1, seg = tid & 1;
    const __bf16* src = A + (size_t)(m0 + row) * K + k0 + seg * 32;
#if USE_ASYNC_LDS
    const unsigned ldst = (unsigned)(uintptr_t)(&As[buf][row][seg * 32]);
    const unsigned long long ga = (unsigned long long)(uintptr_t)src;
    asm volatile("global_load_async_to_lds_b128 %0, %1, off"           :: "v"(ldst), "v"(ga) : "memory");
    asm volatile("global_load_async_to_lds_b128 %0, %1, off offset:16" :: "v"(ldst), "v"(ga) : "memory");
    asm volatile("global_load_async_to_lds_b128 %0, %1, off offset:32" :: "v"(ldst), "v"(ga) : "memory");
    asm volatile("global_load_async_to_lds_b128 %0, %1, off offset:48" :: "v"(ldst), "v"(ga) : "memory");
#else
    const uint4* s4 = (const uint4*)src;
    uint4* dst = (uint4*)(&As[buf][row][seg * 32]);
    dst[0] = s4[0]; dst[1] = s4[1]; dst[2] = s4[2]; dst[3] = s4[3];
#endif
  };
  // B tile: fp32 -> bf16, pack K-pairs, transpose into Bs[n][k]
  auto stageB = [&](int buf, int k0) {
    const int kk2 = (tid & 31) * 2;   // even K index
    const int ng  = tid >> 5;
#pragma unroll
    for (int rep = 0; rep < 2; ++rep) {
      const int nb = ng * 8 + rep * 64;
      const float* r0 = Bw + (size_t)(k0 + kk2) * N + n0 + nb;
      const float* r1 = r0 + N;
      const float4 a0 = ((const float4*)r0)[0], a1 = ((const float4*)r0)[1];
      const float4 b0 = ((const float4*)r1)[0], b1v = ((const float4*)r1)[1];
      const float f0[8] = {a0.x, a0.y, a0.z, a0.w, a1.x, a1.y, a1.z, a1.w};
      const float f1[8] = {b0.x, b0.y, b0.z, b0.w, b1v.x, b1v.y, b1v.z, b1v.w};
#pragma unroll
      for (int i = 0; i < 8; ++i) {
        union { __bf16 hh[2]; unsigned u; } pk;
        pk.hh[0] = (__bf16)f0[i];     // K = kk2
        pk.hh[1] = (__bf16)f1[i];     // K = kk2 + 1
        *(unsigned*)&Bs[buf][nb + i][kk2] = pk.u;
      }
      if (k0 + BK < K)
        __builtin_prefetch((const char*)r0 + (size_t)BK * N * sizeof(float), 0, 1);
    }
  };

  // --- software-pipelined main loop: one barrier per K-step -------------
  stageA(0, 0);
  stageB(0, 0);
  int p = 0;
  for (int k0 = 0; k0 < K; k0 += BK) {
#if USE_ASYNC_LDS
    asm volatile("s_wait_asynccnt 0x0" ::: "memory");
#endif
    __syncthreads();                       // buffer p fully staged
    if (k0 + BK < K) {                     // overlap next-tile staging with WMMA
      stageA(p ^ 1, k0 + BK);
      stageB(p ^ 1, k0 + BK);
    }
#pragma unroll
    for (int ks = 0; ks < BK; ks += 32) {
      // A fragment (16x32): lanes 0-15 -> K 0..7 & 16..23, lanes 16-31 -> K 8..15 & 24..31
      union Frag { v16bf v; uint4 q[2]; };
      Frag af;
      {
        const uint4* pa = (const uint4*)(&As[p][wave * 16 + r][ks]);
        af.q[0] = pa[h];
        af.q[1] = pa[2 + h];
      }
      Frag bfr[8];
#pragma unroll
      for (int tn = 0; tn < 8; ++tn) {
        const uint4* pb = (const uint4*)(&Bs[p][tn * 16 + r][ks + h * 8]);
        bfr[tn].q[0] = pb[0];
        bfr[tn].q[1] = pb[2];
      }
#pragma unroll
      for (int tn = 0; tn < 8; ++tn)
        c[tn] = __builtin_amdgcn_wmma_f32_16x16x32_bf16(
            false, af.v, false, bfr[tn].v, (short)0, c[tn], false, false);
    }
    p ^= 1;
  }

  // --- epilogue: C/D layout: VGPR j -> M = j + 8*h, lane%16 -> N
  const int mbase = m0 + wave * 16;
  float gv[8];
  if (MODE == 2) {
#pragma unroll
    for (int j = 0; j < 8; ++j)
      gv[j] = gate[(size_t)(mbase + j + 8 * h) * NE + expert];
  }
#pragma unroll
  for (int tn = 0; tn < 8; ++tn) {
    const int n = n0 + tn * 16 + r;
    const float bn = bias[n];
#pragma unroll
    for (int j = 0; j < 8; ++j) {
      const int m = mbase + j + 8 * h;
      const float v = c[tn][j] + bn;
      if (MODE == 1) {
        Hout[(size_t)m * N + n] = (__bf16)gelu_f(v);
      } else {
        const float val = v * gv[j];
        float* dst = &Acc[(size_t)m * N + n];
        if (expert == 0) *dst = val;
        else             *dst += val;
      }
    }
  }
}

// ---------------------------------------------------------------- host driver
extern "C" void kernel_launch(void* const* d_in, const int* in_sizes, int n_in,
                              void* d_out, int out_size, void* d_ws, size_t ws_size,
                              hipStream_t stream) {
  const float* x  = (const float*)d_in[0];   // [B,S,D]
  const float* wr = (const float*)d_in[1];   // [L,D,E]
  const float* w1 = (const float*)d_in[2];   // [L,E,D,F]
  const float* b1 = (const float*)d_in[3];   // [L,E,F]
  const float* w2 = (const float*)d_in[4];   // [L,E,F,D]
  const float* b2 = (const float*)d_in[5];   // [L,E,D]
  (void)in_sizes; (void)n_in; (void)out_size; (void)ws_size;

  float* outx = (float*)d_out;               // [B,S,D] running residual state
  float* loss = outx + (size_t)NTOK * DIM_D; // scalar

  const size_t XN_BYTES  = (size_t)NTOK * DIM_D * sizeof(__bf16);
  const size_t H_BYTES   = (size_t)NTOK * DIM_F * sizeof(__bf16);
  const size_t ACC_BYTES = (size_t)NTOK * DIM_D * sizeof(float);
  char* ws = (char*)d_ws;
  __bf16* xn   = (__bf16*)ws;
  __bf16* hbuf = (__bf16*)(ws + XN_BYTES);
  float*  acc  = (float*)(ws + XN_BYTES + H_BYTES);
  float*  gate = (float*)(ws + XN_BYTES + H_BYTES + ACC_BYTES);
  float*  stat = gate + (size_t)NTOK * NE;

  const int NX  = NTOK * DIM_D;
  const int NX4 = NX / 4;
  copy4_kernel<<<(NX4 + 255) / 256, 256, 0, stream>>>((const float4*)x, (float4*)outx, NX4);
  zero_kernel<<<1, 256, 0, stream>>>(loss, 1);

  for (int l = 0; l < NL; ++l) {
    zero_kernel<<<1, 256, 0, stream>>>(stat, 16);
    ln_kernel<<<NTOK, 256, 0, stream>>>(outx, xn);
    router_kernel<<<NTOK / 8, 256, 0, stream>>>(
        xn, wr + (size_t)l * DIM_D * NE, gate, stat);
    loss_kernel<<<1, 32, 0, stream>>>(stat, loss);
    for (int e = 0; e < NE; ++e) {
      const size_t le = (size_t)l * NE + e;
      gemm_kernel<1><<<dim3(DIM_F / 128, NTOK / 128), 256, 0, stream>>>(
          xn, w1 + le * DIM_D * DIM_F, b1 + le * DIM_F, nullptr, e,
          hbuf, nullptr, NTOK, DIM_F, DIM_D);
      gemm_kernel<2><<<dim3(DIM_D / 128, NTOK / 128), 256, 0, stream>>>(
          hbuf, w2 + le * DIM_F * DIM_D, b2 + le * DIM_D, gate, e,
          nullptr, acc, NTOK, DIM_D, DIM_F);
    }
    add4_kernel<<<(NX4 + 255) / 256, 256, 0, stream>>>((float4*)outx, (const float4*)acc, NX4);
  }
}
